// ModelNet_64450279244209
// MI455X (gfx1250) — matrix operand order repaired and checked by reading, not compiled, not run on hardware
//
#include <hip/hip_runtime.h>
#include <hip/hip_bf16.h>
#include <math.h>

// ---------------------------------------------------------------------------
// MI455X / gfx1250: bf16 WMMA LSTM rollout with async global->LDS staging.
// ---------------------------------------------------------------------------

typedef __attribute__((ext_vector_type(16))) __bf16 v16bf;
typedef __attribute__((ext_vector_type(8)))  float  v8f;

#define BATCH   32768
#define EMBD    128
#define HID     256
#define GATES   1024
#define KTOT    512          // [emb(128) | xt_relu(128) | hx(256)]
#define NSTEP   36

#define BM      128          // batch rows per workgroup
#define BN      128          // gate columns per workgroup (= 32 units x 4 gates)
#define BK      128          // K chunk staged in LDS
#define AP      (BK + 8)     // padded LDS row (bf16 elems): 272B stride -> bank stride 4
#define GP      (BN + 4)     // padded gate row (f32): 528B stride -> bank stride 4

#define TILE_BYTES ((size_t)BM * AP * 2)   // 34816 B per A or B tile

union Frag16 { uint4 q[2]; v16bf v; };

__device__ __forceinline__ float sigf(float x) { return 1.0f / (1.0f + __expf(-x)); }

// --- CDNA5 async copy: global memory -> LDS, tracked by ASYNCcnt -----------
__device__ __forceinline__ void async_copy_b128(void* lds_ptr, const void* gptr) {
    uint32_t lds_addr = (uint32_t)(uintptr_t)lds_ptr;
    asm volatile("global_load_async_to_lds_b128 %0, %1, off"
                 :: "v"(lds_addr), "v"(gptr)
                 : "memory");
}
__device__ __forceinline__ void wait_async0() {
    asm volatile("s_wait_asynccnt 0" ::: "memory");
}

// ---------------------------------------------------------------------------
// Prep: Wp[q][k] bf16, q = 128*cb + 32*g + j  <->  orig gate row g*256 + 32*cb + j
//       k<256 from w_ih, k>=256 from w_hh.  biasp[q] = b_ih + b_hh (same perm).
// ---------------------------------------------------------------------------
__global__ void prep_weights_kernel(const float* __restrict__ w_ih,
                                    const float* __restrict__ w_hh,
                                    const float* __restrict__ b_ih,
                                    const float* __restrict__ b_hh,
                                    __bf16* __restrict__ Wp,
                                    float* __restrict__ biasp) {
    int idx = blockIdx.x * 256 + threadIdx.x;          // over 1024*512
    if (idx >= GATES * KTOT) return;
    int q   = idx >> 9;            // permuted gate row
    int k   = idx & (KTOT - 1);
    int cb  = q >> 7;
    int rem = q & 127;
    int g   = rem >> 5;
    int j   = rem & 31;
    int orig = g * 256 + cb * 32 + j;
    float v = (k < HID) ? w_ih[(size_t)orig * HID + k]
                        : w_hh[(size_t)orig * HID + (k - HID)];
    Wp[idx] = (__bf16)v;
    if (k == 0) biasp[q] = b_ih[orig] + b_hh[orig];
}

// ---------------------------------------------------------------------------
// emb = relu(bn1(fc1(x))) -> bf16 [B,128].  One block per batch row.
// ---------------------------------------------------------------------------
__global__ void emb_kernel(const float* __restrict__ x,      // [B,22]
                           const float* __restrict__ fc1_w,  // [128,22]
                           const float* __restrict__ fc1_b,
                           const float* __restrict__ gamma,
                           const float* __restrict__ beta,
                           const float* __restrict__ mean,
                           const float* __restrict__ var,
                           __bf16* __restrict__ embB) {
    __shared__ float xs[22];
    int b = blockIdx.x;
    int c = threadIdx.x;                 // 128 threads
    if (c < 22) xs[c] = x[(size_t)b * 22 + c];
    __syncthreads();
    float s = fc1_b[c];
#pragma unroll
    for (int k = 0; k < 22; ++k) s += xs[k] * fc1_w[c * 22 + k];
    float sc = gamma[c] * rsqrtf(var[c] + 1e-5f);
    s = (s - mean[c]) * sc + beta[c];
    embB[(size_t)b * EMBD + c] = (__bf16)fmaxf(s, 0.0f);
}

// ---------------------------------------------------------------------------
// Fused LSTM step: gates GEMM (bf16 WMMA, K=512) + activations + cell update.
// grid = (8 colblocks, B/128 rowblocks), block = 256 (8 waves).
// Double-buffered LDS tiles, filled by GLOBAL_LOAD_ASYNC_TO_LDS_B128.
// ---------------------------------------------------------------------------
__global__ void
__launch_bounds__(256, 2)
lstm_step_kernel(const __bf16* __restrict__ embB,   // [B,128]
                 const __bf16* __restrict__ hxin,   // [B,256]
                 __bf16*       __restrict__ hxout,  // [B,256]
                 float*        __restrict__ cx,     // [B,256] in-place (per-slice)
                 const __bf16* __restrict__ Wp,     // [1024,512] permuted
                 const float*  __restrict__ biasp,  // [1024]   permuted
                 const float*  __restrict__ fc2_w,  // [128,6]
                 const float*  __restrict__ fc2_b,  // [128]
                 const float*  __restrict__ xt_in)  // prev out [B,6] or nullptr (t==0)
{
    extern __shared__ char smem[];
    __bf16* Ab[2] = { (__bf16*)(smem),
                      (__bf16*)(smem + TILE_BYTES) };
    __bf16* Bb[2] = { (__bf16*)(smem + 2 * TILE_BYTES),
                      (__bf16*)(smem + 3 * TILE_BYTES) };
    float*  Glds  = (float*)smem;                   // [BM][GP] (reuse after GEMM)

    const int tid  = threadIdx.x;
    const int lane = tid & 31;
    const int wave = tid >> 5;
    const int wm   = wave & 1;      // M half
    const int wn   = wave >> 1;     // gate strip 0..3 (i,f,g,o)
    const int cb   = blockIdx.x;    // unit block (32 units)
    const int row0 = blockIdx.y * BM;

    const int n_l  = lane & 15;
    const int khiA = (lane & 16) ? 8  : 0;   // A lane K base offset
    const int khiB = (lane & 16) ? 8  : 0;   // B lane K base offset
    const int mhi  = (lane & 16) ? 8  : 0;   // D lane M offset

    v8f acc[4][2];
#pragma unroll
    for (int mi = 0; mi < 4; ++mi)
#pragma unroll
        for (int ni = 0; ni < 2; ++ni) acc[mi][ni] = (v8f)0.0f;

    // per-thread fc2 row (column fixed by tid for the xt_relu tile)
    const int xc = tid & 127;
    float w2[6];
#pragma unroll
    for (int k = 0; k < 6; ++k) w2[k] = fc2_w[xc * 6 + k];
    const float b2 = fc2_b[xc];

    // ---- async staging helpers (8 x b128 per thread per tile) -------------
    auto stageB = [&](int ch, __bf16* dst) {
#pragma unroll
        for (int i = tid; i < BN * (BK / 8); i += 256) {
            int r = i >> 4, c = i & 15;
            async_copy_b128(dst + r * AP + c * 8,
                            Wp + (size_t)(cb * BN + r) * KTOT + ch * BK + c * 8);
        }
    };
    auto stageA = [&](const __bf16* src, int pitch, int coff, __bf16* dst) {
#pragma unroll
        for (int i = tid; i < BM * (BK / 8); i += 256) {
            int r = i >> 4, c = i & 15;
            async_copy_b128(dst + r * AP + c * 8,
                            src + (size_t)(row0 + r) * pitch + coff + c * 8);
        }
    };

    // ---- prologue: stage chunk 0 (A = emb, B = W[:,0:128]) -----------------
    stageA(embB, EMBD, 0, Ab[0]);
    stageB(0, Bb[0]);
    wait_async0();
    __syncthreads();

#pragma unroll
    for (int ch = 0; ch < 4; ++ch) {
        const int p = ch & 1;
        // ---- stage next chunk into the other buffer while we compute ------
        if (ch == 0) {
            // chunk 1 A-tile: xt_relu computed in place (DS stores, overlaps)
            for (int i = tid; i < BM * BK; i += 256) {
                int r = i >> 7;
                float s = b2;
                if (xt_in) {
                    const float* xr = xt_in + (size_t)(row0 + r) * 6;
#pragma unroll
                    for (int k = 0; k < 6; ++k) s += xr[k] * w2[k];
                } else {
#pragma unroll
                    for (int k = 0; k < 6; ++k) s += -0.5f * w2[k];
                }
                Ab[1][r * AP + xc] = (__bf16)fmaxf(s, 0.0f);
            }
            stageB(1, Bb[1]);
        } else if (ch == 1) {
            stageA(hxin, HID, 0, Ab[0]);       // chunk 2: hx cols 0..127
            stageB(2, Bb[0]);
        } else if (ch == 2) {
            stageA(hxin, HID, BK, Ab[1]);      // chunk 3: hx cols 128..255
            stageB(3, Bb[1]);
        }

        // ---- WMMA over this K chunk (buffers Ab[p]/Bb[p]) ------------------
        const __bf16* Al = Ab[p];
        const __bf16* Bl = Bb[p];
#pragma unroll
        for (int kk = 0; kk < BK; kk += 32) {
            Frag16 fa[4];
#pragma unroll
            for (int mi = 0; mi < 4; ++mi) {
                int m = wm * 64 + mi * 16 + n_l;
                const __bf16* ap = Al + m * AP + kk + khiA;
                fa[mi].q[0] = *(const uint4*)(ap);        // K+0..7
                fa[mi].q[1] = *(const uint4*)(ap + 16);   // K+16..23
            }
            Frag16 fb[2];
#pragma unroll
            for (int ni = 0; ni < 2; ++ni) {
                int n = wn * 32 + ni * 16 + n_l;
                const __bf16* bp = Bl + n * AP + kk + khiB;
                fb[ni].q[0] = *(const uint4*)(bp);        // K+0..7
                fb[ni].q[1] = *(const uint4*)(bp + 16);   // K+16..23
            }
#pragma unroll
            for (int mi = 0; mi < 4; ++mi)
#pragma unroll
                for (int ni = 0; ni < 2; ++ni)
                    acc[mi][ni] = __builtin_amdgcn_wmma_f32_16x16x32_bf16(
                        false, fa[mi].v, false, fb[ni].v,
                        (short)0, acc[mi][ni], false, false);
        }

        if (ch < 3) wait_async0();    // our staging issued this iter is done
        __syncthreads();              // all waves: staging visible, reads done
    }

    // ---- bias + per-gate activation -> LDS (wave-uniform gate type: wn) ----
#pragma unroll
    for (int mi = 0; mi < 4; ++mi) {
#pragma unroll
        for (int ni = 0; ni < 2; ++ni) {
            int n = wn * 32 + ni * 16 + n_l;
            float bias = biasp[cb * BN + n];
#pragma unroll
            for (int r = 0; r < 8; ++r) {
                int m = wm * 64 + mi * 16 + mhi + r;
                float g = acc[mi][ni][r] + bias;
                float a = (wn == 2) ? tanhf(g) : sigf(g);  // g-gate tanh, else sigmoid
                Glds[m * GP + n] = a;
            }
        }
    }
    __syncthreads();

    // ---- LSTM cell update: 128 rows x 32 units -----------------------------
    for (int i = tid; i < BM * 32; i += 256) {
        int r = i >> 5, u = i & 31;
        float iv = Glds[r * GP + u];
        float fv = Glds[r * GP + 32 + u];
        float gv = Glds[r * GP + 64 + u];
        float ov = Glds[r * GP + 96 + u];
        size_t gi = (size_t)(row0 + r) * HID + cb * 32 + u;
        float cn = fv * cx[gi] + iv * gv;
        cx[gi]    = cn;
        hxout[gi] = (__bf16)(ov * tanhf(cn));
    }
}

// ---------------------------------------------------------------------------
// out_t = hx_t @ fc3_w.T + fc3_b  -> d_out[t]  (also next step's xt)
// ---------------------------------------------------------------------------
__global__ void out_proj_kernel(const __bf16* __restrict__ hx,    // [B,256]
                                const float*  __restrict__ fc3_w, // [6,256]
                                const float*  __restrict__ fc3_b,
                                float* __restrict__ out) {        // [B,6]
    int idx = blockIdx.x * 256 + threadIdx.x;
    if (idx >= BATCH * 6) return;
    int b = idx / 6, j = idx - b * 6;
    const __bf16* h = hx + (size_t)b * HID;
    const float*  w = fc3_w + j * HID;
    float s = fc3_b[j];
#pragma unroll 8
    for (int k = 0; k < HID; ++k) s += (float)h[k] * w[k];
    out[idx] = s;
}

// ---------------------------------------------------------------------------
extern "C" void kernel_launch(void* const* d_in, const int* in_sizes, int n_in,
                              void* d_out, int out_size, void* d_ws, size_t ws_size,
                              hipStream_t stream) {
    const float* x      = (const float*)d_in[0];
    const float* fc1_w  = (const float*)d_in[1];
    const float* fc1_b  = (const float*)d_in[2];
    const float* bn_g   = (const float*)d_in[3];
    const float* bn_b   = (const float*)d_in[4];
    const float* bn_m   = (const float*)d_in[5];
    const float* bn_v   = (const float*)d_in[6];
    const float* fc2_w  = (const float*)d_in[7];
    const float* fc2_b  = (const float*)d_in[8];
    const float* fc3_w  = (const float*)d_in[9];
    const float* fc3_b  = (const float*)d_in[10];
    const float* w_ih   = (const float*)d_in[11];
    const float* w_hh   = (const float*)d_in[12];
    const float* b_ih   = (const float*)d_in[13];
    const float* b_hh   = (const float*)d_in[14];
    float* out = (float*)d_out;

    // workspace carve-up (all 256B aligned)
    char* p = (char*)d_ws;
    auto take = [&](size_t bytes) { char* r = p; p += (bytes + 255) & ~(size_t)255; return r; };
    __bf16* Wp    = (__bf16*)take((size_t)GATES * KTOT * 2);   //  1.0 MB
    float*  biasp = (float*) take((size_t)GATES * 4);          //  4 KB
    __bf16* embB  = (__bf16*)take((size_t)BATCH * EMBD * 2);   //  8.4 MB
    __bf16* hxA   = (__bf16*)take((size_t)BATCH * HID * 2);    // 16.8 MB
    __bf16* hxB   = (__bf16*)take((size_t)BATCH * HID * 2);    // 16.8 MB
    float*  cx    = (float*) take((size_t)BATCH * HID * 4);    // 33.6 MB
    (void)ws_size; (void)in_sizes; (void)n_in; (void)out_size;

    prep_weights_kernel<<<(GATES * KTOT + 255) / 256, 256, 0, stream>>>(
        w_ih, w_hh, b_ih, b_hh, Wp, biasp);
    emb_kernel<<<BATCH, 128, 0, stream>>>(x, fc1_w, fc1_b, bn_g, bn_b, bn_m, bn_v, embB);
    hipMemsetAsync(hxA, 0, (size_t)BATCH * HID * 2, stream);   // hx0 = 0 (bf16 zero)
    hipMemsetAsync(cx,  0, (size_t)BATCH * HID * 4, stream);   // cx0 = 0

    const size_t smem = 4 * TILE_BYTES;                        // 139264 B
    __bf16* hbuf[2] = { hxA, hxB };
    for (int t = 0; t < NSTEP; ++t) {
        const float* xt_in = (t == 0) ? nullptr : (out + (size_t)(t - 1) * BATCH * 6);
        lstm_step_kernel<<<dim3(GATES / BN, BATCH / BM), 256, smem, stream>>>(
            embB, hbuf[t & 1], hbuf[(t + 1) & 1], cx, Wp, biasp, fc2_w, fc2_b, xt_in);
        out_proj_kernel<<<(BATCH * 6 + 255) / 256, 256, 0, stream>>>(
            hbuf[(t + 1) & 1], fc3_w, fc3_b, out + (size_t)t * BATCH * 6);
    }
}
